// TDG_Mamba_26938034881207
// MI455X (gfx1250) — matrix operand order repaired and checked by reading, not compiled, not run on hardware
//
#include <hip/hip_runtime.h>
#include <hip/hip_bf16.h>

// ---------------------------------------------------------------------------
// Problem constants (match reference)
// ---------------------------------------------------------------------------
#define NB   1024      // batch / nodes
#define LL   64        // sequence length (== H feature dim of GCN)
#define DM   256       // model dim
#define DI   512       // inner dim
#define DS   16        // state dim
#define DCV  4         // conv width
#define DTR  16        // dt rank
#define HH   64        // GCN hidden
#define GK   DM        // GEMM K
#define GN   (2*DI)    // GEMM N (=1024)

typedef _Float16 h8v  __attribute__((ext_vector_type(8)));
typedef _Float16 v16h __attribute__((ext_vector_type(16)));
typedef float    v8f  __attribute__((ext_vector_type(8)));

union V16U { v16h v; h8v h[2]; };

// ---------------------------------------------------------------------------
// f32 -> f16 conversion
// ---------------------------------------------------------------------------
__global__ void cvt_f16(const float* __restrict__ in, _Float16* __restrict__ out, size_t n) {
  size_t i = (size_t)blockIdx.x * blockDim.x + threadIdx.x;
  if (i < n) out[i] = (_Float16)in[i];
}

// wsum[d] = sum_dm out_proj_w[dm, d]   (out_proj_w is (DM, DI) row-major)
__global__ void colsum_outproj(const float* __restrict__ W, float* __restrict__ wsum) {
  int d = blockIdx.x * blockDim.x + threadIdx.x;
  if (d < DI) {
    float a = 0.f;
    for (int m = 0; m < DM; ++m) a += W[(size_t)m * DI + d];
    wsum[d] = a;
  }
}

// ---------------------------------------------------------------------------
// Big GEMM: xz (M x 1024) = Xf16 (M x 256) @ Wf16^T   (W is (1024,256) row-major)
// WMMA f32_16x16x32_f16.  Block = 4 waves, each wave does a 16x64 strip.
// Register double-buffering over K so next-tile loads overlap current WMMAs.
// ---------------------------------------------------------------------------
__global__ __launch_bounds__(128)
void inproj_gemm(const _Float16* __restrict__ A, const _Float16* __restrict__ B,
                 float* __restrict__ C) {
  const int lane = threadIdx.x;        // 0..31
  const int wave = threadIdx.y;        // 0..3
  const int m0 = (blockIdx.x * 4 + wave) * 16;
  const int n0 = blockIdx.y * 64;
  const int lm = lane & 15;
  const int hi = lane >> 4;

  const _Float16* arow = A + (size_t)(m0 + lm) * GK;
  const _Float16* bcol = B + (size_t)(n0 + lm) * GK;

  v8f acc0 = {}, acc1 = {}, acc2 = {}, acc3 = {};

  // A fragment: row m per lane; lanes 0-15 hold K=k0..k0+15,
  //   lanes 16-31 hold K=k0+16..k0+31 (matches B).
  // B fragments: column n per lane; lanes 0-15 hold K=k0..k0+15,
  //   lanes 16-31 hold K=k0+16..k0+31 (contiguous 16 halves = 32B).
  V16U a, b0, b1, b2, b3;
  {
    a.h[0] = *(const h8v*)(arow + 16 * hi);
    a.h[1] = *(const h8v*)(arow + 16 * hi + 8);
    const _Float16* bb = bcol + 16 * hi;
    b0.h[0] = *(const h8v*)(bb + 0 * 16 * GK); b0.h[1] = *(const h8v*)(bb + 0 * 16 * GK + 8);
    b1.h[0] = *(const h8v*)(bb + 1 * 16 * GK); b1.h[1] = *(const h8v*)(bb + 1 * 16 * GK + 8);
    b2.h[0] = *(const h8v*)(bb + 2 * 16 * GK); b2.h[1] = *(const h8v*)(bb + 2 * 16 * GK + 8);
    b3.h[0] = *(const h8v*)(bb + 3 * 16 * GK); b3.h[1] = *(const h8v*)(bb + 3 * 16 * GK + 8);
  }

#pragma unroll
  for (int k0 = 0; k0 < GK; k0 += 32) {
    // Prefetch next k-step (clamped on last iteration; redundant but in-bounds).
    const int kn = (k0 + 32 < GK) ? (k0 + 32) : k0;
    V16U an, c0, c1, c2, c3;
    an.h[0] = *(const h8v*)(arow + kn + 16 * hi);
    an.h[1] = *(const h8v*)(arow + kn + 16 * hi + 8);
    const _Float16* bn = bcol + kn + 16 * hi;
    c0.h[0] = *(const h8v*)(bn + 0 * 16 * GK); c0.h[1] = *(const h8v*)(bn + 0 * 16 * GK + 8);
    c1.h[0] = *(const h8v*)(bn + 1 * 16 * GK); c1.h[1] = *(const h8v*)(bn + 1 * 16 * GK + 8);
    c2.h[0] = *(const h8v*)(bn + 2 * 16 * GK); c2.h[1] = *(const h8v*)(bn + 2 * 16 * GK + 8);
    c3.h[0] = *(const h8v*)(bn + 3 * 16 * GK); c3.h[1] = *(const h8v*)(bn + 3 * 16 * GK + 8);

    acc0 = __builtin_amdgcn_wmma_f32_16x16x32_f16(false, a.v, false, b0.v, (short)0, acc0, false, false);
    acc1 = __builtin_amdgcn_wmma_f32_16x16x32_f16(false, a.v, false, b1.v, (short)0, acc1, false, false);
    acc2 = __builtin_amdgcn_wmma_f32_16x16x32_f16(false, a.v, false, b2.v, (short)0, acc2, false, false);
    acc3 = __builtin_amdgcn_wmma_f32_16x16x32_f16(false, a.v, false, b3.v, (short)0, acc3, false, false);

    a = an; b0 = c0; b1 = c1; b2 = c2; b3 = c3;
  }

  // C/D layout: VGPR r -> M = r + 8*hi, N = n-tile + lm
  v8f accs[4] = {acc0, acc1, acc2, acc3};
#pragma unroll
  for (int nt = 0; nt < 4; ++nt) {
    int n = n0 + nt * 16 + lm;
#pragma unroll
    for (int r = 0; r < 8; ++r) {
      int m = m0 + r + 8 * hi;
      C[(size_t)m * GN + n] = accs[nt][r];
    }
  }
}

// ---------------------------------------------------------------------------
// Fused bidirectional Mamba tail: conv+silu, x_proj, dt_proj, selective scan,
// gated sum with colsum(out_proj), relu + LayerNorm over L.
// One block (256 threads) per batch row n.  Dynamic LDS ~140KB (CDNA5: 320KB).
// ---------------------------------------------------------------------------
__device__ __forceinline__ float silu_f(float x) { return x / (1.f + __expf(-x)); }

__global__ __launch_bounds__(256)
void mamba_scan(const float* __restrict__ xz,      // (NB, LL, 1024)
                const float* __restrict__ conv_w,  // (DI, DCV)
                const float* __restrict__ conv_b,  // (DI)
                const float* __restrict__ xproj_w, // (48, DI)
                const float* __restrict__ dtw,     // (DI, DTR)
                const float* __restrict__ dtb,     // (DI)
                const float* __restrict__ A_log,   // (DI, DS)
                const float* __restrict__ Dp,      // (DI)
                const float* __restrict__ wsum,    // (DI)
                const float* __restrict__ ln_w,    // (LL)
                const float* __restrict__ ln_b,    // (LL)
                float* __restrict__ s_out) {       // (NB, LL)
  extern __shared__ float sm[];
  float* xc  = sm;                    // LL*DI
  float* dtr = xc + LL * DI;          // LL*DTR
  float* Bt  = dtr + LL * DTR;        // LL*DS
  float* Ct  = Bt + LL * DS;          // LL*DS
  float* sL  = Ct + LL * DS;          // LL
  float* st  = sL + LL;               // 2 (mu, rstd)

  const int n   = blockIdx.x;
  const int tid = threadIdx.x;
  const float* xzn = xz + (size_t)n * LL * GN;

  for (int l = tid; l < LL; l += 256) sL[l] = 0.f;

  // Per-thread channel constants (2 channels: tid, tid+256)
  float Areg[2][DS], dtwreg[2][DTR], cw[2][DCV];
  float dtbr[2], Dr[2], wsr[2], cb[2];
#pragma unroll
  for (int c = 0; c < 2; ++c) {
    int d = tid + 256 * c;
#pragma unroll
    for (int s = 0; s < DS; ++s)  Areg[c][s]   = -__expf(A_log[d * DS + s]);
#pragma unroll
    for (int r = 0; r < DTR; ++r) dtwreg[c][r] = dtw[d * DTR + r];
#pragma unroll
    for (int j = 0; j < DCV; ++j) cw[c][j]     = conv_w[d * DCV + j];
    dtbr[c] = dtb[d]; Dr[c] = Dp[d]; wsr[c] = wsum[d]; cb[c] = conv_b[d];
  }

  for (int dir = 0; dir < 2; ++dir) {
    __syncthreads();
    // causal depthwise conv (over logical/possibly-reversed sequence) + silu
#pragma unroll
    for (int c = 0; c < 2; ++c) {
      int d = tid + 256 * c;
      for (int lp = 0; lp < LL; ++lp) {
        float acc = cb[c];
#pragma unroll
        for (int j = 0; j < DCV; ++j) {
          int lq = lp - (DCV - 1) + j;
          if (lq >= 0) {
            int lo = dir ? (LL - 1 - lq) : lq;
            acc += cw[c][j] * xzn[(size_t)lo * GN + d];
          }
        }
        xc[lp * DI + d] = silu_f(acc);
      }
    }
    __syncthreads();
    // x_db = xc @ x_proj_w^T  (48 outputs per position)
    for (int idx = tid; idx < LL * 48; idx += 256) {
      int lp = idx / 48, r = idx % 48;
      const float* w = xproj_w + (size_t)r * DI;
      const float* xrow = xc + lp * DI;
      float acc = 0.f;
      for (int k = 0; k < DI; ++k) acc += xrow[k] * w[k];
      if (r < DTR)          dtr[lp * DTR + r]       = acc;
      else if (r < DTR+DS)  Bt[lp * DS + (r - DTR)] = acc;
      else                  Ct[lp * DS + (r - DTR - DS)] = acc;
    }
    __syncthreads();
    // selective scan
    float h[2][DS];
#pragma unroll
    for (int c = 0; c < 2; ++c)
#pragma unroll
      for (int s = 0; s < DS; ++s) h[c][s] = 0.f;

    for (int lp = 0; lp < LL; ++lp) {
      int lo = dir ? (LL - 1 - lp) : lp;
      float contrib = 0.f;
#pragma unroll
      for (int c = 0; c < 2; ++c) {
        int d = tid + 256 * c;
        float dtv = dtbr[c];
#pragma unroll
        for (int r = 0; r < DTR; ++r) dtv += dtr[lp * DTR + r] * dtwreg[c][r];
        dtv = (dtv > 20.f) ? dtv : __logf(1.f + __expf(dtv));   // softplus
        float u  = xc[lp * DI + d];
        float du = dtv * u;
        float y  = 0.f;
#pragma unroll
        for (int s = 0; s < DS; ++s) {
          h[c][s] = __expf(dtv * Areg[c][s]) * h[c][s] + du * Bt[lp * DS + s];
          y += h[c][s] * Ct[lp * DS + s];
        }
        float z = xzn[(size_t)lo * GN + DI + d];
        contrib += (y + u * Dr[c]) * silu_f(z) * wsr[c];
      }
      atomicAdd(&sL[lo], contrib);   // ds_add_f32
    }
  }
  __syncthreads();
  // relu + LayerNorm over L
  if (tid == 0) {
    float mu = 0.f;
    for (int l = 0; l < LL; ++l) { float v = fmaxf(sL[l], 0.f); sL[l] = v; mu += v; }
    mu *= (1.f / LL);
    float var = 0.f;
    for (int l = 0; l < LL; ++l) { float d0 = sL[l] - mu; var += d0 * d0; }
    var *= (1.f / LL);
    st[0] = mu; st[1] = rsqrtf(var + 1e-5f);
  }
  __syncthreads();
  if (tid < LL)
    s_out[(size_t)n * LL + tid] = (sL[tid] - st[0]) * st[1] * ln_w[tid] + ln_b[tid];
}

// ---------------------------------------------------------------------------
// GCN helpers
// ---------------------------------------------------------------------------
__global__ void gcn_init(float* dega, float* degb, float* acc_in, float* acc_out) {
  int i = blockIdx.x * blockDim.x + threadIdx.x;
  if (i < NB) { dega[i] = 1.f; degb[i] = 1.f; }          // self-loops
  if (i < NB * HH) { acc_in[i] = 0.f; acc_out[i] = 0.f; }
}

__global__ void gcn_zero_acc(float* acc_in, float* acc_out) {
  int i = blockIdx.x * blockDim.x + threadIdx.x;
  if (i < NB * HH) { acc_in[i] = 0.f; acc_out[i] = 0.f; }
}

__global__ void gcn_deg(const int* __restrict__ ei0, const int* __restrict__ ei1,
                        float* dega, float* degb, int E) {
  int e = blockIdx.x * blockDim.x + threadIdx.x;
  if (e < E) {
    atomicAdd(&dega[ei1[e]], 1.f);   // in-direction: dst = ei[1]
    atomicAdd(&degb[ei0[e]], 1.f);   // out-direction: dst = ei[0]
  }
}

__global__ void gcn_deg_fin(float* dega, float* degb) {
  int i = blockIdx.x * blockDim.x + threadIdx.x;
  if (i < NB) { dega[i] = rsqrtf(dega[i]); degb[i] = rsqrtf(degb[i]); }
}

// hin = X@Wi, hout = X@Wo, root = X@Wr  (X: NB x 64, W: 64 x 64)
__global__ __launch_bounds__(64)
void gcn_mm(const float* __restrict__ X, const float* __restrict__ Wi,
            const float* __restrict__ Wo, const float* __restrict__ Wr,
            float* __restrict__ hi_, float* __restrict__ ho_, float* __restrict__ rt_) {
  __shared__ float xrow[HH];
  int n = blockIdx.x, j = threadIdx.x;
  xrow[j] = X[(size_t)n * HH + j];
  __syncthreads();
  float ai = 0.f, ao = 0.f, ar = 0.f;
  for (int k = 0; k < HH; ++k) {
    float xv = xrow[k];
    ai += xv * Wi[k * HH + j];
    ao += xv * Wo[k * HH + j];
    ar += xv * Wr[k * HH + j];
  }
  hi_[(size_t)n * HH + j] = ai;
  ho_[(size_t)n * HH + j] = ao;
  rt_[(size_t)n * HH + j] = ar;
}

__global__ __launch_bounds__(256)
void gcn_scatter(const int* __restrict__ ei0, const int* __restrict__ ei1,
                 const float* __restrict__ hin, const float* __restrict__ hout,
                 const float* __restrict__ da, const float* __restrict__ db,
                 float* acc_in, float* acc_out, int E) {
  int e = blockIdx.x * blockDim.x + threadIdx.x;
  if (e >= E) return;
  int si = ei0[e], di_ = ei1[e];            // in-conv: src=ei0, dst=ei1
  float ci = da[si] * da[di_];
  int so = ei1[e], do_ = ei0[e];            // out-conv: src=ei1, dst=ei0
  float co = db[so] * db[do_];
  const float* hsi = hin + (size_t)si * HH;
  const float* hso = hout + (size_t)so * HH;
  float* ai = acc_in + (size_t)di_ * HH;
  float* ao = acc_out + (size_t)do_ * HH;
  for (int j = 0; j < HH; ++j) {
    atomicAdd(&ai[j], hsi[j] * ci);
    atomicAdd(&ao[j], hso[j] * co);
  }
}

// out = 0.5*(acc_in + self_in + b_in) + 0.5*(acc_out + self_out + b_out) + root + b_root
__global__ void gcn_combine(const float* acc_in, const float* acc_out,
                            const float* hin, const float* hout, const float* rt,
                            const float* da, const float* db,
                            const float* b_in, const float* b_out, const float* b_root,
                            float* out, int do_relu) {
  int i = blockIdx.x * blockDim.x + threadIdx.x;
  if (i >= NB * HH) return;
  int n = i / HH, j = i % HH;
  float self_in  = hin[i] * da[n] * da[n];
  float self_out = hout[i] * db[n] * db[n];
  float v = 0.5f * (acc_in[i] + self_in + b_in[j])
          + 0.5f * (acc_out[i] + self_out + b_out[j])
          + rt[i] + b_root[j];
  out[i] = do_relu ? fmaxf(v, 0.f) : v;
}

// ---------------------------------------------------------------------------
// Host launcher
// ---------------------------------------------------------------------------
static inline size_t align256(size_t x) { return (x + 255) & ~(size_t)255; }

extern "C" void kernel_launch(void* const* d_in, const int* in_sizes, int n_in,
                              void* d_out, int out_size, void* d_ws, size_t ws_size,
                              hipStream_t stream) {
  const float* x         = (const float*)d_in[0];
  const int*   edge      = (const int*)d_in[3];
  const float* in_proj_w = (const float*)d_in[4];
  const float* conv_w    = (const float*)d_in[5];
  const float* conv_b    = (const float*)d_in[6];
  const float* x_proj_w  = (const float*)d_in[7];
  const float* dt_proj_w = (const float*)d_in[8];
  const float* dt_proj_b = (const float*)d_in[9];
  const float* A_log     = (const float*)d_in[10];
  const float* D_param   = (const float*)d_in[11];
  const float* out_proj_w= (const float*)d_in[12];
  const float* ln_w      = (const float*)d_in[13];
  const float* ln_b      = (const float*)d_in[14];
  const float* c1_in_w   = (const float*)d_in[15];
  const float* c1_in_b   = (const float*)d_in[16];
  const float* c1_out_w  = (const float*)d_in[17];
  const float* c1_out_b  = (const float*)d_in[18];
  const float* c1_root_w = (const float*)d_in[19];
  const float* c1_root_b = (const float*)d_in[20];
  const float* c2_in_w   = (const float*)d_in[21];
  const float* c2_in_b   = (const float*)d_in[22];
  const float* c2_out_w  = (const float*)d_in[23];
  const float* c2_out_b  = (const float*)d_in[24];
  const float* c2_root_w = (const float*)d_in[25];
  const float* c2_root_b = (const float*)d_in[26];

  const int E = in_sizes[3] / 2;
  const int* ei0 = edge;
  const int* ei1 = edge + E;

  const size_t M = (size_t)NB * LL;        // 65536 GEMM rows
  char* ws = (char*)d_ws;
  size_t off = 0;
  _Float16* x_f16 = (_Float16*)(ws + off); off = align256(off + M * GK * 2);
  _Float16* w_f16 = (_Float16*)(ws + off); off = align256(off + (size_t)GN * GK * 2);
  float* xz       = (float*)(ws + off);    off = align256(off + M * GN * 4);
  float* wsum     = (float*)(ws + off);    off = align256(off + DI * 4);
  float* s_buf    = (float*)(ws + off);    off = align256(off + NB * LL * 4);
  float* dinv_a   = (float*)(ws + off);    off = align256(off + NB * 4);
  float* dinv_b   = (float*)(ws + off);    off = align256(off + NB * 4);
  float* h_in     = (float*)(ws + off);    off = align256(off + NB * HH * 4);
  float* h_out    = (float*)(ws + off);    off = align256(off + NB * HH * 4);
  float* rootb    = (float*)(ws + off);    off = align256(off + NB * HH * 4);
  float* acc_in   = (float*)(ws + off);    off = align256(off + NB * HH * 4);
  float* acc_out  = (float*)(ws + off);    off = align256(off + NB * HH * 4);
  float* l1       = (float*)(ws + off);    off = align256(off + NB * HH * 4);

  // 1) precision staging for WMMA
  {
    size_t nx = M * GK;
    cvt_f16<<<dim3((unsigned)((nx + 255) / 256)), dim3(256), 0, stream>>>(x, x_f16, nx);
    size_t nw = (size_t)GN * GK;
    cvt_f16<<<dim3((unsigned)((nw + 255) / 256)), dim3(256), 0, stream>>>(in_proj_w, w_f16, nw);
  }
  colsum_outproj<<<dim3(2), dim3(256), 0, stream>>>(out_proj_w, wsum);

  // 2) big GEMM: xz = x @ in_proj_w^T (shared by both mamba directions)
  inproj_gemm<<<dim3((unsigned)(M / 64), GN / 64), dim3(32, 4), 0, stream>>>(x_f16, w_f16, xz);

  // 3) fused bidirectional mamba tail -> normalized s (NB, LL)
  {
    size_t smem = (size_t)(LL * DI + LL * DTR + 2 * LL * DS + LL + 8) * sizeof(float);
    mamba_scan<<<dim3(NB), dim3(256), smem, stream>>>(
        xz, conv_w, conv_b, x_proj_w, dt_proj_w, dt_proj_b,
        A_log, D_param, wsum, ln_w, ln_b, s_buf);
  }

  // 4) GCN degrees (shared by both layers)
  gcn_init<<<dim3((NB * HH + 255) / 256), dim3(256), 0, stream>>>(dinv_a, dinv_b, acc_in, acc_out);
  gcn_deg<<<dim3((E + 255) / 256), dim3(256), 0, stream>>>(ei0, ei1, dinv_a, dinv_b, E);
  gcn_deg_fin<<<dim3((NB + 255) / 256), dim3(256), 0, stream>>>(dinv_a, dinv_b);

  // 5) layer 1
  gcn_mm<<<dim3(NB), dim3(HH), 0, stream>>>(s_buf, c1_in_w, c1_out_w, c1_root_w, h_in, h_out, rootb);
  gcn_scatter<<<dim3((E + 255) / 256), dim3(256), 0, stream>>>(ei0, ei1, h_in, h_out,
                                                               dinv_a, dinv_b, acc_in, acc_out, E);
  gcn_combine<<<dim3((NB * HH + 255) / 256), dim3(256), 0, stream>>>(
      acc_in, acc_out, h_in, h_out, rootb, dinv_a, dinv_b,
      c1_in_b, c1_out_b, c1_root_b, l1, 1);

  // 6) layer 2 -> d_out
  gcn_zero_acc<<<dim3((NB * HH + 255) / 256), dim3(256), 0, stream>>>(acc_in, acc_out);
  gcn_mm<<<dim3(NB), dim3(HH), 0, stream>>>(l1, c2_in_w, c2_out_w, c2_root_w, h_in, h_out, rootb);
  gcn_scatter<<<dim3((E + 255) / 256), dim3(256), 0, stream>>>(ei0, ei1, h_in, h_out,
                                                               dinv_a, dinv_b, acc_in, acc_out, E);
  gcn_combine<<<dim3((NB * HH + 255) / 256), dim3(256), 0, stream>>>(
      acc_in, acc_out, h_in, h_out, rootb, dinv_a, dinv_b,
      c2_in_b, c2_out_b, c2_root_b, (float*)d_out, 0);

  (void)n_in; (void)out_size; (void)ws_size; (void)off;
}
